// EncoderTransformerBlock_43130061586465
// MI455X (gfx1250) — matrix operand order repaired and checked by reading, not compiled, not run on hardware
//
#include <hip/hip_runtime.h>
#include <math.h>

typedef __attribute__((ext_vector_type(16))) _Float16 v16h;
typedef __attribute__((ext_vector_type(8)))  float    v8f;
typedef __attribute__((ext_vector_type(4)))  uint32_t u32x4;
typedef __attribute__((ext_vector_type(4)))  float    f32x4;
typedef _Float16 f16;

#define LANE ((int)(threadIdx.x & 31))
#define WAVE ((int)(threadIdx.x >> 5))

// ---------------------------------------------------------------- helpers ---

__device__ __forceinline__ v8f vzero8() {
  v8f z;
#pragma unroll
  for (int i = 0; i < 8; ++i) z[i] = 0.f;
  return z;
}

__device__ __forceinline__ v8f wmma32(v16h a, v16h b, v8f c) {
  // D = A(16x32 f16) * B(32x16 f16) + C(16x16 f32)
  return __builtin_amdgcn_wmma_f32_16x16x32_f16(false, a, false, b, (short)0, c,
                                                false, false);
}

union HF16 {
  v16h v;
  u32x4 q[2];
};

// A fragment: 16x32 f16 from row-major source (contiguous in K).
// Lane l: row M = m0 + (l&15), K-half (l>>4)*8; two b128 loads.
__device__ __forceinline__ v16h frag_a(const f16* s, int ld, int m0, int k0) {
  int l = LANE;
  const f16* p = s + (size_t)(m0 + (l & 15)) * ld + k0 + ((l >> 4) << 3);
  HF16 a;
  a.q[0] = *(const u32x4*)p;
  a.q[1] = *(const u32x4*)(p + 16);
  return a.v;
}

// B fragment: 32x16 f16 where storage is transposed (s[n*ld + k]).
// Lane l: col N = n0 + (l&15), K-half (l>>4)*8; two b128 loads (K, K+16).
__device__ __forceinline__ v16h frag_bt(const f16* s, int ld, int k0, int n0) {
  int l = LANE;
  const f16* p = s + (size_t)(n0 + (l & 15)) * ld + k0 + ((l >> 4) << 3);
  HF16 b;
  b.q[0] = *(const u32x4*)p;
  b.q[1] = *(const u32x4*)(p + 16);
  return b.v;
}

// C/D fragment element r: row M = r + 8*lane[4], col N = lane&15.
__device__ __forceinline__ int cm(int r) {
  return r + (((threadIdx.x >> 4) & 1) << 3);
}
__device__ __forceinline__ int cn() { return (int)(threadIdx.x & 15); }

__device__ __forceinline__ float wred_sum(float v) {
#pragma unroll
  for (int o = 16; o >= 1; o >>= 1) v += __shfl_xor(v, o, 32);
  return v;
}
__device__ __forceinline__ float wred_max(float v) {
#pragma unroll
  for (int o = 16; o >= 1; o >>= 1) v = fmaxf(v, __shfl_xor(v, o, 32));
  return v;
}

// Async global->LDS 16B copy (CDNA5 GLOBAL_LOAD_ASYNC_TO_LDS_B128, ASYNCcnt).
__device__ __forceinline__ void async_cp16(uint32_t lds_off, const void* g) {
  asm volatile("global_load_async_to_lds_b128 %0, %1, off" ::"v"(lds_off),
               "v"((uint64_t)(uintptr_t)g)
               : "memory");
}
__device__ __forceinline__ void async_wait0() {
  asm volatile("s_wait_asynccnt 0x0" ::: "memory");
}
__device__ __forceinline__ uint32_t lds_off_of(const void* p) {
  return (uint32_t)(uintptr_t)p;  // LDS aperture: low 32 bits = LDS offset
}

// ---------------------------------------------------------------- constants -

constexpr int   C     = 256;
constexpr int   NWIN  = 2048;  // 2 * 32 * 32 windows
constexpr int   NT    = 64;    // tokens / window
constexpr int   NFP   = 96;    // 66 spectral rows padded to 96 (3 K-steps)
constexpr int   NFR   = 66;
constexpr float SCALE = 0.17677669529663687f;  // 32^-0.5

// ------------------------------------------------------------------ kernels -

// Convert weights to f16, build DFT/iDFT matrices and expanded RPB bias.
__global__ void k_prep(const float* qkvw, const float* projw, const float* fc1w,
                       const float* fc2w, const float* rpb, f16* qkvw_h,
                       f16* projw_h, f16* fc1w_h, f16* fc2w_h, f16* Fm, f16* Fi,
                       float* biasx) {
  const int stride = gridDim.x * blockDim.x;
  const int tid = blockIdx.x * blockDim.x + threadIdx.x;
  const float PI2_64 = 6.283185307179586f / 64.f;
  for (int i = tid; i < 768 * 256; i += stride) qkvw_h[i] = (f16)qkvw[i];
  for (int i = tid; i < 256 * 256; i += stride) projw_h[i] = (f16)projw[i];
  for (int i = tid; i < 1024 * 256; i += stride) fc1w_h[i] = (f16)fc1w[i];
  for (int i = tid; i < 256 * 1024; i += stride) fc2w_h[i] = (f16)fc2w[i];
  // Forward rfft matrix (ortho): [96][64]; rows 0..32 real, 33..65 imag.
  for (int i = tid; i < NFP * NT; i += stride) {
    int r = i / NT, n = i % NT;
    float v = 0.f;
    if (r < 33) {
      v = cosf(PI2_64 * (float)((r * n) & 63)) * 0.125f;
    } else if (r < NFR) {
      int kk = r - 33;
      v = -sinf(PI2_64 * (float)((kk * n) & 63)) * 0.125f;
    }
    Fm[i] = (f16)v;
  }
  // Inverse rfft matrix (ortho): [64][96], cols 66..95 zero.
  for (int i = tid; i < NT * NFP; i += stride) {
    int n = i / NFP, j = i % NFP;
    float v = 0.f;
    if (j < 33) {
      float cc = (j == 0 || j == 32) ? 1.f : 2.f;
      v = cc * cosf(PI2_64 * (float)((j * n) & 63)) * 0.125f;
    } else if (j < NFR) {
      int kk = j - 33;
      float cc = (kk == 0 || kk == 32) ? 0.f : 2.f;  // irfft drops DC/Nyq imag
      v = -cc * sinf(PI2_64 * (float)((kk * n) & 63)) * 0.125f;
    }
    Fi[i] = (f16)v;
  }
  // Expanded relative-position bias: [8][64][64]
  for (int i = tid; i < 8 * 64 * 64; i += stride) {
    int h = i >> 12, ii = (i >> 6) & 63, jj = i & 63;
    int dy = (ii >> 3) - (jj >> 3) + 7;
    int dx = (ii & 7) - (jj & 7) + 7;
    biasx[i] = rpb[(dy * 15 + dx) * 8 + h];
  }
}

// LayerNorm1 + window partition, f32 -> f16; writes [tok][C] and [C][tok].
// Lane owns 8 contiguous channels: vector loads + one packed b128 store.
__global__ void k_ln1(const float* __restrict__ x, const float* w,
                      const float* b, f16* xw, f16* xwT) {
  int t = blockIdx.x * 8 + WAVE;  // 0..131071
  int l = LANE;
  const float* src = x + (size_t)t * C + l * 8;
  f32x4 v0 = ((const f32x4*)src)[0];
  f32x4 v1 = ((const f32x4*)src)[1];
  float s = 0.f, s2 = 0.f;
#pragma unroll
  for (int j = 0; j < 4; ++j) {
    s += v0[j] + v1[j];
    s2 += v0[j] * v0[j] + v1[j] * v1[j];
  }
  s = wred_sum(s);
  s2 = wred_sum(s2);
  float mean = s * (1.f / 256.f);
  float var = s2 * (1.f / 256.f) - mean * mean;
  float rstd = rsqrtf(var + 1e-5f);
  int bb = t >> 16, hh = (t >> 8) & 255, ww = t & 255;
  int win = bb * 1024 + (hh >> 3) * 32 + (ww >> 3);
  int tw = ((hh & 7) << 3) + (ww & 7);
  f16* dst = xw + ((size_t)win * NT + tw) * C;
  f16* dstT = xwT + (size_t)win * C * NT + tw;
  union {
    u32x4 q;
    f16 h[8];
  } pk;
#pragma unroll
  for (int j = 0; j < 8; ++j) {
    int c = l * 8 + j;
    float f = (j < 4) ? v0[j] : v1[j - 4];
    f16 y = (f16)((f - mean) * rstd * w[c] + b[c]);
    pk.h[j] = y;
    dstT[(size_t)c * NT] = y;
  }
  *(u32x4*)(dst + l * 8) = pk.q;
}

// Spectral transform xf = F @ xw per window (WMMA, M=96 K=64 N=256).
__global__ void k_dft(const f16* __restrict__ xwT_g,
                      const f16* __restrict__ F_g, f16* __restrict__ xf_g) {
  extern __shared__ char smem[];
  f16* XT = (f16*)smem;            // [256][64]  32768
  f16* Fl = (f16*)(smem + 32768);  // [96][64]   12288
  int win = blockIdx.x;
  {
    const char* g = (const char*)(xwT_g + (size_t)win * C * NT);
    uint32_t l0 = lds_off_of(XT);
    for (int i = threadIdx.x; i < 2048; i += 256)  // 32KB
      async_cp16(l0 + i * 16, g + i * 16);
    const char* gf = (const char*)F_g;
    uint32_t l1 = lds_off_of(Fl);
    for (int i = threadIdx.x; i < 768; i += 256)  // 12KB
      async_cp16(l1 + i * 16, gf + i * 16);
    async_wait0();
  }
  __syncthreads();
  f16* dst = xf_g + (size_t)win * NFP * C;
  for (int t = WAVE; t < 96; t += 8) {  // 6 Mt x 16 Nt
    int mt = t >> 4, nt = t & 15;
    v8f acc = vzero8();
    for (int k0 = 0; k0 < 64; k0 += 32)
      acc = wmma32(frag_a(Fl, 64, mt * 16, k0), frag_bt(XT, NT, k0, nt * 16),
                   acc);
    int n = nt * 16 + cn();
#pragma unroll
    for (int r = 0; r < 8; ++r)
      dst[(size_t)(mt * 16 + cm(r)) * C + n] = (f16)acc[r];
  }
}

// Stage the per-head, per-kstep weight tile [96][32] via async DMA.
__device__ __forceinline__ void stage_w(f16* Wst, const f16* qkvw_h, int h,
                                        int ks) {
  uint32_t wl = lds_off_of(Wst);
  for (int i = threadIdx.x; i < 768; i += 256) {  // 768 x 16B = 12KB
    int rr = i >> 2, cc = i & 3;
    int s = rr >> 5, d = rr & 31;
    const f16* g =
        qkvw_h + (size_t)(s * 256 + h * 32 + d) * C + ks * 32 + cc * 8;
    async_cp16(wl + i * 16, g);
  }
  async_wait0();
}

// Spatial window attention: per-block window, per-head loop, all WMMA.
__global__ void k_attn_sp(const f16* __restrict__ xw_g,
                          const f16* __restrict__ qkvw_h,
                          const float* __restrict__ qkv_b,
                          const float* __restrict__ biasx,
                          f16* __restrict__ outsp) {
  extern __shared__ char smem[];
  f16* X = (f16*)smem;                 // [64][256]              32768
  f16* QKV = (f16*)(smem + 32768);     // Q[64][32] K[64][32] Vt[32][64]
  float* SC = (float*)(smem + 45056);  // [64][64] f32           16384
  f16* Wst = (f16*)(smem + 45056);     // [96][32] f16           (overlay)
  f16* P = (f16*)(smem + 61440);       // [64][64] f16            8192
  int win = blockIdx.x;
  {
    const char* g = (const char*)(xw_g + (size_t)win * NT * C);
    uint32_t l0 = lds_off_of(X);
    for (int i = threadIdx.x; i < 2048; i += 256)  // 32KB
      async_cp16(l0 + i * 16, g + i * 16);
    async_wait0();
  }
  __syncthreads();
  f16* Q = QKV;
  f16* K = QKV + NT * 32;
  f16* Vt = QKV + 2 * NT * 32;  // [32][64]
  for (int h = 0; h < 8; ++h) {
    // ---- fused QKV GEMM for this head: (64x256) @ (256x96) ----
    v8f acc[3];
#pragma unroll
    for (int i = 0; i < 3; ++i) acc[i] = vzero8();
    for (int ks = 0; ks < 8; ++ks) {
      stage_w(Wst, qkvw_h, h, ks);
      __syncthreads();
      int i = 0;
      for (int t = WAVE; t < 24; t += 8, ++i) {  // 4 Mt x 6 Nt
        int mt = t / 6, nt = t % 6;
        acc[i] = wmma32(frag_a(X, C, mt * 16, ks * 32),
                        frag_bt(Wst, 32, 0, nt * 16), acc[i]);
      }
      __syncthreads();
    }
    {
      int i = 0;
      for (int t = WAVE; t < 24; t += 8, ++i) {
        int mt = t / 6, nt = t % 6;
        int rr = nt * 16 + cn();
        int s = nt >> 1;  // tile-uniform
        int d = rr & 31;
        float qb = qkv_b[s * 256 + h * 32 + d];
        if (s < 2) {
#pragma unroll
          for (int r = 0; r < 8; ++r)
            QKV[((size_t)s * NT + mt * 16 + cm(r)) * 32 + d] =
                (f16)(acc[i][r] + qb);
        } else {  // V stored transposed for the PV GEMM
#pragma unroll
          for (int r = 0; r < 8; ++r)
            Vt[(size_t)d * NT + mt * 16 + cm(r)] = (f16)(acc[i][r] + qb);
        }
      }
    }
    __syncthreads();
    // ---- S = Q K^T * scale + bias (K=32: single WMMA per tile) ----
    for (int t = WAVE; t < 16; t += 8) {
      int mt = t >> 2, nt = t & 3;
      v8f s8 = wmma32(frag_a(Q, 32, mt * 16, 0), frag_bt(K, 32, 0, nt * 16),
                      vzero8());
      int n = nt * 16 + cn();
#pragma unroll
      for (int r = 0; r < 8; ++r) {
        int m = mt * 16 + cm(r);
        SC[m * 64 + n] = s8[r] * SCALE + biasx[((size_t)h * 64 + m) * 64 + n];
      }
    }
    __syncthreads();
    // ---- softmax (8 rows per wave) ----
    {
      int l = LANE;
      for (int rrow = WAVE * 8; rrow < WAVE * 8 + 8; ++rrow) {
        float a0 = SC[rrow * 64 + l];
        float a1 = SC[rrow * 64 + l + 32];
        float mx = wred_max(fmaxf(a0, a1));
        float e0 = __expf(a0 - mx), e1 = __expf(a1 - mx);
        float inv = 1.f / wred_sum(e0 + e1);
        P[rrow * 64 + l] = (f16)(e0 * inv);
        P[rrow * 64 + l + 32] = (f16)(e1 * inv);
      }
    }
    __syncthreads();
    // ---- O = P @ V ----
    for (int t = WAVE; t < 8; t += 8) {
      int mt = t >> 1, nt = t & 1;
      v8f o = vzero8();
      for (int k0 = 0; k0 < NT; k0 += 32)
        o = wmma32(frag_a(P, 64, mt * 16, k0), frag_bt(Vt, NT, k0, nt * 16), o);
      int n = nt * 16 + cn();
#pragma unroll
      for (int r = 0; r < 8; ++r)
        outsp[((size_t)win * NT + mt * 16 + cm(r)) * C + h * 32 + n] =
            (f16)o[r];
    }
    __syncthreads();
  }
}

// Spectral attention on the 66 (padded 96) rfft rows; output transposed.
__global__ void k_attn_f(const f16* __restrict__ xf_g,
                         const f16* __restrict__ qkvw_h,
                         const float* __restrict__ qkv_b,
                         f16* __restrict__ outfT) {
  extern __shared__ char smem[];
  f16* X = (f16*)smem;                 // [96][256]             49152
  f16* QKV = (f16*)(smem + 49152);     // Q[96][32] K[96][32] Vt[32][96]
  float* SC = (float*)(smem + 67584);  // [96][96] f32          36864
  f16* Wst = (f16*)(smem + 67584);     // [96][32]              (overlay)
  f16* P = (f16*)(smem + 104448);      // [96][96] f16          18432
  int win = blockIdx.x;
  {
    const char* g = (const char*)(xf_g + (size_t)win * NFP * C);
    uint32_t l0 = lds_off_of(X);
    for (int i = threadIdx.x; i < 3072; i += 256)  // 48KB
      async_cp16(l0 + i * 16, g + i * 16);
    async_wait0();
  }
  __syncthreads();
  f16* Q = QKV;
  f16* K = QKV + NFP * 32;
  f16* Vt = QKV + 2 * NFP * 32;  // [32][96]
  for (int h = 0; h < 8; ++h) {
    v8f acc[5];
#pragma unroll
    for (int i = 0; i < 5; ++i) acc[i] = vzero8();
    for (int ks = 0; ks < 8; ++ks) {
      stage_w(Wst, qkvw_h, h, ks);
      __syncthreads();
      int i = 0;
      for (int t = WAVE; t < 36; t += 8, ++i) {  // 6 Mt x 6 Nt
        int mt = t / 6, nt = t % 6;
        acc[i] = wmma32(frag_a(X, C, mt * 16, ks * 32),
                        frag_bt(Wst, 32, 0, nt * 16), acc[i]);
      }
      __syncthreads();
    }
    {
      int i = 0;
      for (int t = WAVE; t < 36; t += 8, ++i) {
        int mt = t / 6, nt = t % 6;
        int rr = nt * 16 + cn();
        int s = nt >> 1;
        int d = rr & 31;
        float qb = qkv_b[s * 256 + h * 32 + d];
        if (s < 2) {
#pragma unroll
          for (int r = 0; r < 8; ++r)
            QKV[((size_t)s * NFP + mt * 16 + cm(r)) * 32 + d] =
                (f16)(acc[i][r] + qb);
        } else {
#pragma unroll
          for (int r = 0; r < 8; ++r)
            Vt[(size_t)d * NFP + mt * 16 + cm(r)] = (f16)(acc[i][r] + qb);
        }
      }
    }
    __syncthreads();
    for (int t = WAVE; t < 36; t += 8) {  // S: 6x6 tiles
      int mt = t / 6, nt = t % 6;
      v8f s8 = wmma32(frag_a(Q, 32, mt * 16, 0), frag_bt(K, 32, 0, nt * 16),
                      vzero8());
      int n = nt * 16 + cn();
#pragma unroll
      for (int r = 0; r < 8; ++r)
        SC[(mt * 16 + cm(r)) * NFP + n] = s8[r] * SCALE;
    }
    __syncthreads();
    // softmax over valid 66 cols; rows >= 66 zeroed (keeps K-padding clean)
    {
      int l = LANE;
      for (int rrow = WAVE * 12; rrow < WAVE * 12 + 12; ++rrow) {
        float a0 = SC[rrow * NFP + l];
        float a1 = SC[rrow * NFP + l + 32];
        int n3 = l + 64;
        float a2 = (n3 < NFR) ? SC[rrow * NFP + n3] : -1e30f;
        float mx = wred_max(fmaxf(a0, fmaxf(a1, a2)));
        float e0 = __expf(a0 - mx), e1 = __expf(a1 - mx);
        float e2 = (n3 < NFR) ? __expf(a2 - mx) : 0.f;
        float sum = wred_sum(e0 + e1 + e2);
        float inv = (rrow < NFR) ? 1.f / sum : 0.f;
        P[rrow * NFP + l] = (f16)(e0 * inv);
        P[rrow * NFP + l + 32] = (f16)(e1 * inv);
        P[rrow * NFP + n3] = (f16)(e2 * inv);
      }
    }
    __syncthreads();
    for (int t = WAVE; t < 12; t += 8) {  // O: 6 Mt x 2 Nt
      int mt = t >> 1, nt = t & 1;
      v8f o = vzero8();
      for (int k0 = 0; k0 < NFP; k0 += 32)
        o = wmma32(frag_a(P, NFP, mt * 16, k0), frag_bt(Vt, NFP, k0, nt * 16),
                   o);
      int n = nt * 16 + cn();
      // transposed store: outfT[win][c][freq], contiguous in freq
      f16* drow = outfT + ((size_t)win * C + h * 32 + n) * NFP + mt * 16;
#pragma unroll
      for (int r = 0; r < 8; ++r) drow[cm(r)] = (f16)o[r];
    }
    __syncthreads();
  }
}

// xr = Finv @ outf; add spatial branch; proj GEMM + shortcut; un-window.
__global__ void k_proj(const f16* __restrict__ outfT_g,
                       const f16* __restrict__ outsp_g,
                       const f16* __restrict__ Fi_g,
                       const f16* __restrict__ projw_h,
                       const float* __restrict__ proj_b,
                       const float* __restrict__ x0, float* __restrict__ xres) {
  extern __shared__ char smem[];
  f16* S = (f16*)smem;              // [64][256] 32768
  f16* Fi = (f16*)(smem + 32768);   // [64][96]  12288
  f16* OFt = (f16*)(smem + 45056);  // [256][96] 49152
  int win = blockIdx.x;
  {
    uint32_t l0 = lds_off_of(Fi);
    const char* gf = (const char*)Fi_g;
    for (int i = threadIdx.x; i < 768; i += 256)  // 12KB
      async_cp16(l0 + i * 16, gf + i * 16);
    uint32_t l1 = lds_off_of(OFt);
    const char* go = (const char*)(outfT_g + (size_t)win * C * NFP);
    for (int i = threadIdx.x; i < 3072; i += 256)  // 48KB
      async_cp16(l1 + i * 16, go + i * 16);
    async_wait0();
  }
  __syncthreads();
  const f16* osp = outsp_g + (size_t)win * NT * C;
  for (int t = WAVE; t < 64; t += 8) {  // 4 Mt x 16 Nt
    int mt = t >> 4, nt = t & 15;
    v8f a = vzero8();
    for (int k0 = 0; k0 < NFP; k0 += 32)
      a = wmma32(frag_a(Fi, NFP, mt * 16, k0), frag_bt(OFt, NFP, k0, nt * 16),
                 a);
    int n = nt * 16 + cn();
#pragma unroll
    for (int r = 0; r < 8; ++r) {
      int m = mt * 16 + cm(r);
      S[m * C + n] = (f16)(a[r] + (float)osp[(size_t)m * C + n]);
    }
  }
  __syncthreads();
  int bb = win >> 10, wr = (win >> 5) & 31, wc = win & 31;
  for (int t = WAVE; t < 64; t += 8) {
    int mt = t >> 4, nt = t & 15;
    v8f a = vzero8();
    for (int k0 = 0; k0 < C; k0 += 32)
      a = wmma32(frag_a(S, C, mt * 16, k0), frag_bt(projw_h, C, k0, nt * 16),
                 a);
    int n = nt * 16 + cn();
    float pb = proj_b[n];
#pragma unroll
    for (int r = 0; r < 8; ++r) {
      int m = mt * 16 + cm(r);
      int hh = wr * 8 + (m >> 3), ww = wc * 8 + (m & 7);
      size_t idx = (((size_t)bb * 256 + hh) * 256 + ww) * C + n;
      xres[idx] = x0[idx] + a[r] + pb;
    }
  }
}

// LN2 + fc1 GEMM + exact GELU -> f16 hidden.
__global__ void k_mlp1(const float* __restrict__ xres, const float* w2,
                       const float* b2, const f16* __restrict__ fc1w_h,
                       const float* __restrict__ fc1_b, f16* __restrict__ h1) {
  extern __shared__ char smem[];
  f16* A = (f16*)smem;  // [64][256]
  size_t tok0 = (size_t)blockIdx.x * 64;
  {
    int l = LANE;
    for (int i = WAVE; i < 64; i += 8) {
      const float* src = xres + (tok0 + i) * C + l * 8;
      f32x4 v0 = ((const f32x4*)src)[0];
      f32x4 v1 = ((const f32x4*)src)[1];
      float s = 0.f, s2 = 0.f;
#pragma unroll
      for (int j = 0; j < 4; ++j) {
        s += v0[j] + v1[j];
        s2 += v0[j] * v0[j] + v1[j] * v1[j];
      }
      s = wred_sum(s);
      s2 = wred_sum(s2);
      float mean = s * (1.f / 256.f);
      float var = s2 * (1.f / 256.f) - mean * mean;
      float rstd = rsqrtf(var + 1e-5f);
      union {
        u32x4 q;
        f16 h[8];
      } pk;
#pragma unroll
      for (int j = 0; j < 8; ++j) {
        int c = l * 8 + j;
        float f = (j < 4) ? v0[j] : v1[j - 4];
        pk.h[j] = (f16)((f - mean) * rstd * w2[c] + b2[c]);
      }
      *(u32x4*)(A + i * C + l * 8) = pk.q;
    }
  }
  __syncthreads();
  for (int t = WAVE; t < 256; t += 8) {  // 4 Mt x 64 Nt
    int mt = t & 3, nt = t >> 2;
    __builtin_prefetch(fc1w_h + (size_t)(nt * 16) * C, 0, 0);
    v8f a = vzero8();
    for (int k0 = 0; k0 < C; k0 += 32)
      a = wmma32(frag_a(A, C, mt * 16, k0), frag_bt(fc1w_h, C, k0, nt * 16), a);
    int n = nt * 16 + cn();
    float fb = fc1_b[n];
#pragma unroll
    for (int r = 0; r < 8; ++r) {
      int m = mt * 16 + cm(r);
      float y = a[r] + fb;
      y = 0.5f * y * (1.f + erff(y * 0.70710678118f));
      h1[(tok0 + m) * 1024 + n] = (f16)y;
    }
  }
}

// fc2 GEMM + residual -> f32 output.
__global__ void k_mlp2(const f16* __restrict__ h1,
                       const f16* __restrict__ fc2w_h,
                       const float* __restrict__ fc2_b,
                       const float* __restrict__ xres,
                       float* __restrict__ out) {
  extern __shared__ char smem[];
  f16* A = (f16*)smem;  // [64][128] chunk
  size_t tok0 = (size_t)blockIdx.x * 64;
  v8f acc[8];
#pragma unroll
  for (int i = 0; i < 8; ++i) acc[i] = vzero8();
  for (int kc = 0; kc < 8; ++kc) {
    {
      uint32_t l0 = lds_off_of(A);
      for (int i = threadIdx.x; i < 1024; i += 256) {  // 16KB chunk
        int row = i >> 4, cp = i & 15;
        const f16* g = h1 + (tok0 + row) * 1024 + kc * 128 + cp * 8;
        async_cp16(l0 + i * 16, g);
      }
      async_wait0();
    }
    __syncthreads();
    int i = 0;
    for (int t = WAVE; t < 64; t += 8, ++i) {  // 4 Mt x 16 Nt
      int mt = t >> 4, nt = t & 15;
      __builtin_prefetch(fc2w_h + (size_t)(nt * 16) * 1024 + kc * 128, 0, 0);
      for (int k0 = 0; k0 < 128; k0 += 32)
        acc[i] = wmma32(frag_a(A, 128, mt * 16, k0),
                        frag_bt(fc2w_h, 1024, kc * 128 + k0, nt * 16), acc[i]);
    }
    __syncthreads();
  }
  int i = 0;
  for (int t = WAVE; t < 64; t += 8, ++i) {
    int mt = t >> 4, nt = t & 15;
    int n = nt * 16 + cn();
    float fb = fc2_b[n];
#pragma unroll
    for (int r = 0; r < 8; ++r) {
      int m = mt * 16 + cm(r);
      size_t idx = (tok0 + m) * C + n;
      out[idx] = xres[idx] + acc[i][r] + fb;
    }
  }
}

// ------------------------------------------------------------------ launch --

extern "C" void kernel_launch(void* const* d_in, const int* in_sizes, int n_in,
                              void* d_out, int out_size, void* d_ws,
                              size_t ws_size, hipStream_t stream) {
  const float* x = (const float*)d_in[0];
  // d_in[1] = mask_matrix (unused: shift_size == 0)
  const float* n1w = (const float*)d_in[2];
  const float* n1b = (const float*)d_in[3];
  const float* qkvw = (const float*)d_in[4];
  const float* qkvb = (const float*)d_in[5];
  const float* rpb = (const float*)d_in[6];
  const float* projw = (const float*)d_in[7];
  const float* projb = (const float*)d_in[8];
  const float* n2w = (const float*)d_in[9];
  const float* n2b = (const float*)d_in[10];
  const float* fc1w = (const float*)d_in[11];
  const float* fc1b = (const float*)d_in[12];
  const float* fc2w = (const float*)d_in[13];
  const float* fc2b = (const float*)d_in[14];

  char* ws = (char*)d_ws;
  size_t off = 0;
  auto alloc = [&](size_t bytes) {
    size_t o = off;
    off = (off + bytes + 255) & ~(size_t)255;
    return o;
  };
  f16* qkvw_h = (f16*)(ws + alloc(768 * 256 * 2));
  f16* projw_h = (f16*)(ws + alloc(256 * 256 * 2));
  f16* fc1w_h = (f16*)(ws + alloc(1024 * 256 * 2));
  f16* fc2w_h = (f16*)(ws + alloc(256 * 1024 * 2));
  f16* Fm = (f16*)(ws + alloc((size_t)NFP * NT * 2));
  f16* Fi = (f16*)(ws + alloc((size_t)NT * NFP * 2));
  float* biasx = (float*)(ws + alloc(8 * 64 * 64 * 4));
  f16* xw = (f16*)(ws + alloc((size_t)NWIN * NT * C * 2));
  f16* xwT = (f16*)(ws + alloc((size_t)NWIN * C * NT * 2));
  f16* xf = (f16*)(ws + alloc((size_t)NWIN * NFP * C * 2));
  f16* outsp = (f16*)(ws + alloc((size_t)NWIN * NT * C * 2));
  f16* outfT = (f16*)(ws + alloc((size_t)NWIN * C * NFP * 2));
  float* xres = (float*)(ws + alloc((size_t)NWIN * NT * C * 4));
  f16* h1 = (f16*)(ws + alloc((size_t)NWIN * NT * 1024 * 2));

  k_prep<<<512, 256, 0, stream>>>(qkvw, projw, fc1w, fc2w, rpb, qkvw_h, projw_h,
                                  fc1w_h, fc2w_h, Fm, Fi, biasx);
  k_ln1<<<16384, 256, 0, stream>>>(x, n1w, n1b, xw, xwT);
  k_dft<<<NWIN, 256, 45056, stream>>>(xwT, Fm, xf);
  k_attn_sp<<<NWIN, 256, 69632, stream>>>(xw, qkvw_h, qkvb, biasx, outsp);
  k_attn_f<<<NWIN, 256, 122880, stream>>>(xf, qkvw_h, qkvb, outfT);
  k_proj<<<NWIN, 256, 94208, stream>>>(outfT, outsp, Fi, projw_h, projb, x,
                                       xres);
  k_mlp1<<<NWIN, 256, 32768, stream>>>(xres, n2w, n2b, fc1w_h, fc1b, h1);
  k_mlp2<<<NWIN, 256, 16384, stream>>>(h1, fc2w_h, fc2b, xres, (float*)d_out);
}
